// LSTM_AE_74972949119044
// MI455X (gfx1250) — compile-verified, matrix operand order checked by reading, not run on hardware
//
#include <hip/hip_runtime.h>
#include <hip/hip_bf16.h>

// ---------------------------------------------------------------------------
// LSTM autoencoder for MI455X (gfx1250, wave32, WMMA).
//
// Whh kept f16-resident in LDS across all 8192 recurrent steps (320KB/WGP);
// recurrent matvec and the time-parallel input GEMMs both run on
// v_wmma_f32_16x16x32_f16; Whh preload uses async global->LDS copies
// (ASYNCcnt); cross-WGP h exchange via L2 atomics with release/acquire;
// gates in f32 with xg prefetched ahead of the matvec.
// ---------------------------------------------------------------------------

typedef __attribute__((ext_vector_type(16))) _Float16 v16h;
typedef __attribute__((ext_vector_type(8)))  _Float16 v8h;
typedef __attribute__((ext_vector_type(8)))  float    v8f;

#define SEQ_T 8192

__device__ __forceinline__ float sigmoidf_(float x) {
    return 1.0f / (1.0f + __expf(-x));
}

// ---------------------------------------------------------------------------
// Prep kernels
// ---------------------------------------------------------------------------

__global__ void cvt_f32_to_f16(const float* __restrict__ src,
                               _Float16* __restrict__ dst, int n) {
    int i = blockIdx.x * blockDim.x + threadIdx.x;
    for (; i < n; i += gridDim.x * blockDim.x) dst[i] = (_Float16)src[i];
}

// Pack Wih [N x K] f32 into the per-lane B-operand layout of
// v_wmma_f32_16x16x32_f16: tile (kt, nt) -> 32 lanes x 16 halves contiguous.
__global__ void pack_wihT(const float* __restrict__ wih, _Float16* __restrict__ dst,
                          int K, int N) {
    int total = K * N;
    int o = blockIdx.x * blockDim.x + threadIdx.x;
    for (; o < total; o += gridDim.x * blockDim.x) {
        int j    = o & 15;
        int L    = (o >> 4) & 31;
        int tile = o >> 9;
        int ntile = N >> 4;
        int nt = tile % ntile;
        int kt = tile / ntile;
        int half = L >> 4, c = L & 15;
        int k = kt * 32 + (j < 8 ? half * 8 + j : 16 + half * 8 + (j - 8));
        int n = nt * 16 + c;
        dst[o] = (_Float16)wih[(size_t)n * K + k];
    }
}

// e1 input projection: in_dim == 1 -> xg[t][j] = x[t]*Wih[j] + bih[j] + bhh[j]
__global__ void e1_xg_kernel(const float* __restrict__ x, const float* __restrict__ w,
                             const float* __restrict__ bih, const float* __restrict__ bhh,
                             float* __restrict__ xg) {
    int i = blockIdx.x * blockDim.x + threadIdx.x;
    int total = SEQ_T * 2048;
    for (; i < total; i += gridDim.x * blockDim.x) {
        int t = i >> 11, j = i & 2047;
        xg[i] = x[t] * w[j] + bih[j] + bhh[j];
    }
}

// d1 input projection: decoder input is the repeated latent z, one row for all t.
__global__ void d1_row_kernel(const _Float16* __restrict__ z, const float* __restrict__ wih,
                              const float* __restrict__ bih, const float* __restrict__ bhh,
                              float* __restrict__ row) {
    int j = threadIdx.x;           // 512 threads, N = 512, K = 128
    if (j < 512) {
        float s = 0.f;
        for (int k = 0; k < 128; ++k) s += (float)z[k] * wih[(size_t)j * 128 + k];
        row[j] = s + bih[j] + bhh[j];
    }
}

// ---------------------------------------------------------------------------
// Time-parallel input GEMM:  XG[T x N] f32 = Act[T x K] f16 @ Bpacked + bias
// ---------------------------------------------------------------------------
__global__ __launch_bounds__(256)
void gemm_xg_kernel(const _Float16* __restrict__ act, const _Float16* __restrict__ bp,
                    const float* __restrict__ bih, const float* __restrict__ bhh,
                    float* __restrict__ xg, int K, int N) {
    const int wave = threadIdx.x >> 5;
    const int lane = threadIdx.x & 31;
    const int half = lane >> 4;
    const int c    = lane & 15;

    const int id = blockIdx.x * 8 + wave;       // grid sized exactly
    const int tm = id & 511;                    // T/16 == 512
    const int tn = id >> 9;
    const int m0 = tm << 4, n0 = tn << 4;

    const _Float16* arow = act + (size_t)(m0 + c) * K;

    // split-K into two accumulator chains to hide WMMA latency
    v8f acc0 = {}, acc1 = {};
    const int ntile = N >> 4;
    for (int k0 = 0; k0 < K; k0 += 32) {
        v16h av, bv;
        *(v8h*)&av       = *(const v8h*)(arow + k0 + half * 8);
        *((v8h*)&av + 1) = *(const v8h*)(arow + k0 + 16 + half * 8);
        const _Float16* bptr = bp + (((size_t)(k0 >> 5) * ntile + tn) << 9) + (lane << 4);
        *(v8h*)&bv       = *(const v8h*)(bptr);
        *((v8h*)&bv + 1) = *(const v8h*)(bptr + 8);
        if ((k0 >> 5) & 1)
            acc1 = __builtin_amdgcn_wmma_f32_16x16x32_f16(false, av, false, bv,
                                                          (short)0, acc1, false, false);
        else
            acc0 = __builtin_amdgcn_wmma_f32_16x16x32_f16(false, av, false, bv,
                                                          (short)0, acc0, false, false);
    }

    const float bias = bih[n0 + c] + bhh[n0 + c];
#pragma unroll
    for (int r = 0; r < 8; ++r)
        xg[(size_t)(m0 + r + 8 * half) * N + n0 + c] = acc0[r] + acc1[r] + bias;
}

// ---------------------------------------------------------------------------
// Persistent recurrence kernel.
//   H  : hidden size, HC : h-slice per workgroup, NWG : workgroups.
// Whh f16 slice resident in LDS (+8-half row pad -> 16 A-row readers hit
// 16 distinct bank groups); 8192-step loop; L2 counter barrier across WGs.
// ---------------------------------------------------------------------------
template <int H, int HC, int NWG>
__global__ __launch_bounds__(512)
void rec_kernel(const _Float16* __restrict__ whh16, const float* __restrict__ xg,
                long long xgStride, _Float16* __restrict__ ys,
                _Float16* __restrict__ hbuf, unsigned int* __restrict__ counter) {
    constexpr int ROWS = 4 * HC;        // g-rows handled by this WG
    constexpr int HP   = H + 8;         // padded LDS row stride (halves)
    constexpr int NT   = ROWS / 16;     // 16x16 M-tiles
    constexpr int TPW  = NT / 16;       // tiles per wave (16 waves)

    extern __shared__ __align__(16) char smem[];
    _Float16* lds_whh = (_Float16*)smem;
    _Float16* lds_h   = lds_whh + (size_t)ROWS * HP;
    float*    lds_g   = (float*)(lds_h + H);
    float*    lds_c   = lds_g + ROWS;

    const int tid  = threadIdx.x;
    const int wave = tid >> 5;
    const int lane = tid & 31;
    const int half = lane >> 4;
    const int mrel = lane & 15;
    const int wg   = blockIdx.x;
    const int off  = wg * HC;

    // ---- async preload of the Whh slice (gather 4 gate bands) into LDS ----
    // ASYNCcnt-tracked global->LDS b128 copies (CDNA5 async memory path).
    {
        const unsigned sbase = (unsigned)(size_t)(void*)lds_whh; // low 32b = DS offset
        constexpr int CPR = H / 8;                               // 16B chunks per row
        for (int i = tid; i < ROWS * CPR; i += 512) {
            int r = i / CPR, cc = i % CPR;
            int gate = r / HC, idx = r % HC;
            int grow = gate * H + off + idx;
            unsigned laddr = sbase + (unsigned)(r * HP + cc * 8) * 2u;
            unsigned long long gaddr =
                (unsigned long long)(const void*)(whh16 + (size_t)grow * H + cc * 8);
            asm volatile("global_load_async_to_lds_b128 %0, %1, off"
                         :: "v"(laddr), "v"(gaddr) : "memory");
        }
    }
    for (int i = tid; i < H; i += 512) lds_h[i] = (_Float16)0.f;
    for (int i = tid; i < HC; i += 512) lds_c[i] = 0.f;
    asm volatile("s_wait_asynccnt 0x0" ::: "memory");
    __syncthreads();

    for (int t = 0; t < SEQ_T; ++t) {
        // ---- prefetch this step's gate inputs (independent of h) ----
        float xval = 0.f;
        if (tid < ROWS) {
            const int gate = tid / HC, idx = tid % HC;
            xval = xg[(size_t)t * xgStride + gate * H + off + idx];
        }

        // ---- matvec g = Whh_slice @ h via WMMA ----
        // k0 outer so the B (h) chunk is loaded once per step and reused by
        // all of this wave's tiles; 2 accumulator chains break the RAW chain.
        v8f acc[TPW][2];
#pragma unroll
        for (int ti = 0; ti < TPW; ++ti) { acc[ti][0] = (v8f){}; acc[ti][1] = (v8f){}; }
#pragma unroll
        for (int k0 = 0; k0 < H; k0 += 32) {
            v16h bv;
            *(v8h*)&bv       = *(const v8h*)(lds_h + k0 + half * 8);
            *((v8h*)&bv + 1) = *(const v8h*)(lds_h + k0 + 16 + half * 8);
            const int p = (k0 >> 5) & 1;
#pragma unroll
            for (int ti = 0; ti < TPW; ++ti) {
                const int m0 = (wave + 16 * ti) << 4;
                const _Float16* arow = lds_whh + (size_t)(m0 + mrel) * HP;
                v16h av;
                *(v8h*)&av       = *(const v8h*)(arow + k0 + half * 8);
                *((v8h*)&av + 1) = *(const v8h*)(arow + k0 + 16 + half * 8);
                acc[ti][p] = __builtin_amdgcn_wmma_f32_16x16x32_f16(
                    false, av, false, bv, (short)0, acc[ti][p], false, false);
            }
        }
#pragma unroll
        for (int ti = 0; ti < TPW; ++ti) {
            if (mrel == 0) {                 // lanes 0 and 16 hold column 0
                const int m0 = (wave + 16 * ti) << 4;
#pragma unroll
                for (int r = 0; r < 8; ++r)
                    lds_g[m0 + half * 8 + r] = acc[ti][0][r] + acc[ti][1][r];
            }
        }
        __syncthreads();

        // ---- gate stage 1: nonlinearities, 4*HC-way parallel ----
        if (tid < ROWS) {
            float a = lds_g[tid] + xval;
            const int gate = tid / HC;
            a = (gate == 2) ? tanhf(a) : sigmoidf_(a);
            lds_g[tid] = a;
        }
        __syncthreads();

        // ---- gate stage 2: state update ----
        if (tid < HC) {
            float i_ = lds_g[tid];
            float f_ = lds_g[HC + tid];
            float g_ = lds_g[2 * HC + tid];
            float o_ = lds_g[3 * HC + tid];
            float cn = f_ * lds_c[tid] + i_ * g_;
            lds_c[tid] = cn;
            float hn = o_ * tanhf(cn);
            _Float16 h16 = (_Float16)hn;
            lds_h[off + tid] = h16;
            ys[(size_t)t * H + off + tid] = h16;   // activations stored f16
            if constexpr (NWG > 1) hbuf[off + tid] = h16;
        }

        if constexpr (NWG > 1) {
            __threadfence();                       // flush h-slice stores to L2
            __syncthreads();
            if (tid == 0)
                __hip_atomic_fetch_add(counter, 1u, __ATOMIC_RELAXED,
                                       __HIP_MEMORY_SCOPE_AGENT);
            const unsigned target = (unsigned)(t + 1) * NWG;
            while (__hip_atomic_load(counter, __ATOMIC_ACQUIRE,
                                     __HIP_MEMORY_SCOPE_AGENT) < target)
                __builtin_amdgcn_s_sleep(1);
            for (int i = tid; i < H; i += 512) lds_h[i] = hbuf[i];
            __syncthreads();
        } else {
            __syncthreads();
        }
    }
}

// ---------------------------------------------------------------------------
// Output projection: out[t] = ys[t] (f16, 512) . out_W + out_b
// ---------------------------------------------------------------------------
__global__ __launch_bounds__(64)
void out_kernel(const _Float16* __restrict__ ys, const float* __restrict__ w,
                const float* __restrict__ b, float* __restrict__ out) {
    __shared__ float red[64];
    const int t = blockIdx.x, tid = threadIdx.x;
    float s = 0.f;
    for (int k = tid; k < 512; k += 64) s += (float)ys[(size_t)t * 512 + k] * w[k];
    red[tid] = s;
    __syncthreads();
    for (int o = 32; o > 0; o >>= 1) {
        if (tid < o) red[tid] += red[tid + o];
        __syncthreads();
    }
    if (tid == 0) out[t] = red[0] + b[0];
}

// ---------------------------------------------------------------------------
// Launch
// ---------------------------------------------------------------------------
static inline size_t alignup(size_t x) { return (x + 255) & ~(size_t)255; }

extern "C" void kernel_launch(void* const* d_in, const int* in_sizes, int n_in,
                              void* d_out, int out_size, void* d_ws, size_t ws_size,
                              hipStream_t stream) {
    (void)in_sizes; (void)n_in; (void)out_size; (void)ws_size;

    const float* x = (const float*)d_in[0];
    const float* e1W = (const float*)d_in[1];  const float* e1U = (const float*)d_in[2];
    const float* e1bi = (const float*)d_in[3]; const float* e1bh = (const float*)d_in[4];
    const float* e2W = (const float*)d_in[5];  const float* e2U = (const float*)d_in[6];
    const float* e2bi = (const float*)d_in[7]; const float* e2bh = (const float*)d_in[8];
    const float* e3W = (const float*)d_in[9];  const float* e3U = (const float*)d_in[10];
    const float* e3bi = (const float*)d_in[11];const float* e3bh = (const float*)d_in[12];
    const float* d1W = (const float*)d_in[13]; const float* d1U = (const float*)d_in[14];
    const float* d1bi = (const float*)d_in[15];const float* d1bh = (const float*)d_in[16];
    const float* d2W = (const float*)d_in[17]; const float* d2U = (const float*)d_in[18];
    const float* d2bi = (const float*)d_in[19];const float* d2bh = (const float*)d_in[20];
    const float* d3W = (const float*)d_in[21]; const float* d3U = (const float*)d_in[22];
    const float* d3bi = (const float*)d_in[23];const float* d3bh = (const float*)d_in[24];
    const float* outW = (const float*)d_in[25];
    const float* outB = (const float*)d_in[26];

    // ---- workspace carve-up ----
    char* ws = (char*)d_ws;
    size_t off = 0;
    float*    xgbuf = (float*)(ws + off);    off = alignup(off + (size_t)SEQ_T * 2048 * 4);
    _Float16* ysA   = (_Float16*)(ws + off); off = alignup(off + (size_t)SEQ_T * 512 * 2);
    _Float16* ysB   = (_Float16*)(ws + off); off = alignup(off + (size_t)SEQ_T * 512 * 2);
    _Float16* u16_e1 = (_Float16*)(ws + off); off = alignup(off + (size_t)2048 * 512 * 2);
    _Float16* u16_e2 = (_Float16*)(ws + off); off = alignup(off + (size_t)1024 * 256 * 2);
    _Float16* u16_e3 = (_Float16*)(ws + off); off = alignup(off + (size_t)512 * 128 * 2);
    _Float16* u16_d1 = (_Float16*)(ws + off); off = alignup(off + (size_t)512 * 128 * 2);
    _Float16* u16_d2 = (_Float16*)(ws + off); off = alignup(off + (size_t)1024 * 256 * 2);
    _Float16* u16_d3 = (_Float16*)(ws + off); off = alignup(off + (size_t)2048 * 512 * 2);
    _Float16* wp_e2 = (_Float16*)(ws + off); off = alignup(off + (size_t)512 * 1024 * 2);
    _Float16* wp_e3 = (_Float16*)(ws + off); off = alignup(off + (size_t)256 * 512 * 2);
    _Float16* wp_d2 = (_Float16*)(ws + off); off = alignup(off + (size_t)128 * 1024 * 2);
    _Float16* wp_d3 = (_Float16*)(ws + off); off = alignup(off + (size_t)256 * 2048 * 2);
    float*    xgrow = (float*)(ws + off);    off = alignup(off + 512 * 4);
    _Float16* hbuf  = (_Float16*)(ws + off); off = alignup(off + 512 * 2);
    unsigned* cnt   = (unsigned*)(ws + off); off = alignup(off + 256);

    auto ldsBytes = [](int H, int HC) {
        return (size_t)4 * HC * (H + 8) * 2 + (size_t)H * 2 + (size_t)4 * HC * 4 + (size_t)HC * 4;
    };
    const size_t lds512 = ldsBytes(512, 64);
    const size_t lds256 = ldsBytes(256, 64);
    const size_t lds128 = ldsBytes(128, 128);

    // ---- prep: Whh -> f16, Wih -> packed f16 B operand ----
    cvt_f32_to_f16<<<1024, 256, 0, stream>>>(e1U, u16_e1, 2048 * 512);
    cvt_f32_to_f16<<<512, 256, 0, stream>>>(e2U, u16_e2, 1024 * 256);
    cvt_f32_to_f16<<<256, 256, 0, stream>>>(e3U, u16_e3, 512 * 128);
    cvt_f32_to_f16<<<256, 256, 0, stream>>>(d1U, u16_d1, 512 * 128);
    cvt_f32_to_f16<<<512, 256, 0, stream>>>(d2U, u16_d2, 1024 * 256);
    cvt_f32_to_f16<<<1024, 256, 0, stream>>>(d3U, u16_d3, 2048 * 512);
    pack_wihT<<<1024, 256, 0, stream>>>(e2W, wp_e2, 512, 1024);
    pack_wihT<<<512, 256, 0, stream>>>(e3W, wp_e3, 256, 512);
    pack_wihT<<<512, 256, 0, stream>>>(d2W, wp_d2, 128, 1024);
    pack_wihT<<<1024, 256, 0, stream>>>(d3W, wp_d3, 256, 2048);

    // ---- encoder ----
    e1_xg_kernel<<<(SEQ_T * 2048) / 256, 256, 0, stream>>>(x, e1W, e1bi, e1bh, xgbuf);
    hipMemsetAsync(cnt, 0, sizeof(unsigned), stream);
    rec_kernel<512, 64, 8><<<8, 512, lds512, stream>>>(u16_e1, xgbuf, 2048, ysA, hbuf, cnt);

    gemm_xg_kernel<<<4096, 256, 0, stream>>>(ysA, wp_e2, e2bi, e2bh, xgbuf, 512, 1024);
    hipMemsetAsync(cnt, 0, sizeof(unsigned), stream);
    rec_kernel<256, 64, 4><<<4, 512, lds256, stream>>>(u16_e2, xgbuf, 1024, ysB, hbuf, cnt);

    gemm_xg_kernel<<<2048, 256, 0, stream>>>(ysB, wp_e3, e3bi, e3bh, xgbuf, 256, 512);
    rec_kernel<128, 128, 1><<<1, 512, lds128, stream>>>(u16_e3, xgbuf, 512, ysA, nullptr, nullptr);

    // ---- decoder (latent z = last e3 hidden state, repeated over time) ----
    d1_row_kernel<<<1, 512, 0, stream>>>(ysA + (size_t)(SEQ_T - 1) * 128, d1W, d1bi, d1bh, xgrow);
    rec_kernel<128, 128, 1><<<1, 512, lds128, stream>>>(u16_d1, xgrow, 0, ysB, nullptr, nullptr);

    gemm_xg_kernel<<<4096, 256, 0, stream>>>(ysB, wp_d2, d2bi, d2bh, xgbuf, 128, 1024);
    hipMemsetAsync(cnt, 0, sizeof(unsigned), stream);
    rec_kernel<256, 64, 4><<<4, 512, lds256, stream>>>(u16_d2, xgbuf, 1024, ysA, hbuf, cnt);

    gemm_xg_kernel<<<8192, 256, 0, stream>>>(ysA, wp_d3, d3bi, d3bh, xgbuf, 256, 2048);
    hipMemsetAsync(cnt, 0, sizeof(unsigned), stream);
    rec_kernel<512, 64, 8><<<8, 512, lds512, stream>>>(u16_d3, xgbuf, 2048, ysB, hbuf, cnt);

    out_kernel<<<SEQ_T, 64, 0, stream>>>(ysB, outW, outB, (float*)d_out);
}
